// S4Layer_35459249995880
// MI455X (gfx1250) — matrix operand order repaired and checked by reading, not compiled, not run on hardware
//
#include <hip/hip_runtime.h>
#include <hip/hip_bf16.h>

typedef __bf16 bf16_t;
typedef __attribute__((ext_vector_type(16))) __bf16 v16bf;
typedef __attribute__((ext_vector_type(8)))  __bf16 v8bf;
typedef __attribute__((ext_vector_type(8)))  float  v8f;

#define HAS_ASYNC_LDS __has_builtin(__builtin_amdgcn_global_load_async_to_lds_b128)

// ---------------------------------------------------------------------------
// WMMA fragment helpers (CDNA5 16x16x32 bf16, wave32; A and B use contiguous k = 16*(lane/16) + s)
// ---------------------------------------------------------------------------

__device__ __forceinline__ v16bf frag_pair(const bf16_t* p0, const bf16_t* p1) {
    v8bf lo = *(const v8bf*)p0;
    v8bf hi = *(const v8bf*)p1;
    return __builtin_shufflevector(lo, hi, 0,1,2,3,4,5,6,7,8,9,10,11,12,13,14,15);
}

// A fragment: 16x32 tile of row-major [M x ld] bf16.
__device__ __forceinline__ v16bf load_a(const bf16_t* A, int ld, int m0, int k0, int lane) {
    int hf = lane >> 4;
    const bf16_t* base = A + (size_t)(m0 + (lane & 15)) * ld + k0 + 16 * hf;
    return frag_pair(base, base + 8);
}

// B fragment straight from global: B[k][n] = W[n][k], W row-major [N x ld].
__device__ __forceinline__ v16bf load_b(const bf16_t* W, int ld, int n0, int k0, int lane) {
    int hf = lane >> 4;
    const bf16_t* base = W + (size_t)(n0 + (lane & 15)) * ld + k0 + 16 * hf;
    return frag_pair(base, base + 8);
}

__device__ __forceinline__ v8f wmma_bf16(v16bf a, v16bf b, v8f c) {
    return __builtin_amdgcn_wmma_f32_16x16x32_bf16(false, a, false, b, (short)0, c,
                                                   false, false);
}

__device__ __forceinline__ v8f vzero8() {
    v8f z = {0.f, 0.f, 0.f, 0.f, 0.f, 0.f, 0.f, 0.f};
    return z;
}

// ---------------------------------------------------------------------------
// Async global -> LDS staging (CDNA5 GLOBAL_LOAD_ASYNC_TO_LDS_B128, ASYNCcnt).
// The builtin takes (v4i AS1*, v4i AS3*, imm offset, imm cpol) where v4i is
// int __attribute__((vector_size(16))) -- confirmed by the round-2 diagnostic.
// ---------------------------------------------------------------------------
typedef int b128v_t __attribute__((vector_size(16)));
typedef __attribute__((address_space(1))) b128v_t gb128_t;
typedef __attribute__((address_space(3))) b128v_t lb128_t;

__device__ __forceinline__ void async_copy_b128(const bf16_t* g, bf16_t* lds) {
#if HAS_ASYNC_LDS
    __builtin_amdgcn_global_load_async_to_lds_b128(
        (gb128_t*)(b128v_t*)const_cast<bf16_t*>(g),
        (lb128_t*)(b128v_t*)lds, 0, 0);
#else
    *(v8bf*)lds = *(const v8bf*)g;
#endif
}

__device__ __forceinline__ void stage_wait() {
#if HAS_ASYNC_LDS
#if __has_builtin(__builtin_amdgcn_s_wait_asynccnt)
    __builtin_amdgcn_s_wait_asynccnt(0);
#else
    asm volatile("s_wait_asynccnt 0x0" ::: "memory");
#endif
#endif
    __syncthreads();
}

// ---------------------------------------------------------------------------
// 64-row x 512-col B slice in LDS, 16-byte chunks XOR-swizzled so that the
// 16 lanes of a B-fragment (rows n..n+15, same k) hit distinct bank groups.
// Exactly 64 KB of LDS.
// ---------------------------------------------------------------------------
__device__ __forceinline__ int bchunk_off(int n, int j) {   // element offset
    return (n << 9) + (((j ^ (n & 15)) & 63) << 3);
}

// Wg: 64-row slice of row-major [N x 512] weights.
__device__ __forceinline__ void stage_b512(const bf16_t* Wg, bf16_t* Blds) {
    for (int c = threadIdx.x; c < 64 * 64; c += 256) {   // 64 rows x 64 chunks
        int n = c >> 6, j = c & 63;
        async_copy_b128(Wg + ((size_t)n << 9) + (j << 3), Blds + bchunk_off(n, j));
    }
}

// B fragment from the swizzled LDS slice (n0 relative to slice, 0..48).
__device__ __forceinline__ v16bf load_b_sw(const bf16_t* Blds, int n0, int k0, int lane) {
    int hf = lane >> 4;
    int n  = n0 + (lane & 15);
    int j0 = (k0 >> 3) + 2 * hf;
    const bf16_t* p0 = Blds + bchunk_off(n, j0);
    const bf16_t* p1 = Blds + bchunk_off(n, j0 + 1);
    return frag_pair(p0, p1);
}

// Core K=512 loop: A from global, B from staged LDS slice; 4 accumulators.
__device__ __forceinline__ void gemm_core512(const bf16_t* __restrict__ Ain,
                                             const bf16_t* Blds, int m0, int lane,
                                             v8f acc[4]) {
    for (int k0 = 0; k0 < 512; k0 += 32) {
        v16bf a = load_a(Ain, 512, m0, k0, lane);
#pragma unroll
        for (int j = 0; j < 4; ++j) {
            v16bf b = load_b_sw(Blds, 16 * j, k0, lane);
            acc[j] = wmma_bf16(a, b, acc[j]);
        }
    }
}

// ---------------------------------------------------------------------------
// fp32 -> bf16 convert
// ---------------------------------------------------------------------------
__global__ void __launch_bounds__(256) cvt_bf16(const float* __restrict__ in,
                                                bf16_t* __restrict__ out, int n) {
    int i = blockIdx.x * 256 + threadIdx.x;
    if (i < n) out[i] = (bf16_t)in[i];
}

// ---------------------------------------------------------------------------
// Out[M x 512] = A[M x 512] @ W[512 x 512]^T + bias (fp32 out).
// Block 256 thr = 8 waves covers 128 rows x 64 cols; B slice staged in LDS.
// grid = (M/128, 512/64)
// ---------------------------------------------------------------------------
__global__ void __launch_bounds__(256) gemm_n512(const bf16_t* __restrict__ Ain,
                                                 const bf16_t* __restrict__ W,
                                                 const float* __restrict__ bias,
                                                 float* __restrict__ Out) {
    __shared__ bf16_t Blds[64 * 512];
    int lane  = threadIdx.x & 31;
    int wave  = threadIdx.x >> 5;
    int hf    = lane >> 4;
    int m0    = blockIdx.x * 128 + wave * 16;
    int n0blk = blockIdx.y * 64;

    stage_b512(W + (size_t)n0blk * 512, Blds);
    stage_wait();

    v8f acc[4];
#pragma unroll
    for (int j = 0; j < 4; ++j) acc[j] = vzero8();

    gemm_core512(Ain, Blds, m0, lane, acc);

#pragma unroll
    for (int j = 0; j < 4; ++j) {
        int n = n0blk + 16 * j + (lane & 15);
        float bv = bias[n];
#pragma unroll
        for (int r = 0; r < 8; ++r) {
            int m = m0 + r + 8 * hf;
            Out[(size_t)m * 512 + n] = acc[j][r] + bv;
        }
    }
}

// ---------------------------------------------------------------------------
// Row LayerNorm over 512 cols, bf16 out. One block (256 thr) per row.
// ---------------------------------------------------------------------------
__global__ void __launch_bounds__(256) ln_rows(const float* __restrict__ H,
                                               const float* __restrict__ g,
                                               const float* __restrict__ be,
                                               bf16_t* __restrict__ Hn) {
    __shared__ float r1[256];
    __shared__ float r2[256];
    int tid = threadIdx.x;
    const float* hr = H + (size_t)blockIdx.x * 512;
    float x0 = hr[tid], x1 = hr[tid + 256];
    r1[tid] = x0 + x1;
    r2[tid] = x0 * x0 + x1 * x1;
    __syncthreads();
    for (int s = 128; s > 0; s >>= 1) {
        if (tid < s) { r1[tid] += r1[tid + s]; r2[tid] += r2[tid + s]; }
        __syncthreads();
    }
    float mu  = r1[0] * (1.0f / 512.0f);
    float var = r2[0] * (1.0f / 512.0f) - mu * mu;
    float rs  = rsqrtf(var + 1e-5f);
    bf16_t* o = Hn + (size_t)blockIdx.x * 512;
    o[tid]       = (bf16_t)((x0 - mu) * rs * g[tid] + be[tid]);
    o[tid + 256] = (bf16_t)((x1 - mu) * rs * g[tid + 256] + be[tid + 256]);
}

// ---------------------------------------------------------------------------
// U[M x 64] = Hn[M x 512] @ Bm[64 x 512]^T (fp32 out).
// Block covers 128 rows x all 64 cols; Bm staged in LDS once. grid = M/128.
// ---------------------------------------------------------------------------
__global__ void __launch_bounds__(256) gemm_u(const bf16_t* __restrict__ Hn,
                                              const bf16_t* __restrict__ Bw,
                                              float* __restrict__ U) {
    __shared__ bf16_t Blds[64 * 512];
    int lane = threadIdx.x & 31;
    int wave = threadIdx.x >> 5;
    int hf   = lane >> 4;
    int m0   = blockIdx.x * 128 + wave * 16;

    stage_b512(Bw, Blds);
    stage_wait();

    v8f acc[4];
#pragma unroll
    for (int j = 0; j < 4; ++j) acc[j] = vzero8();

    gemm_core512(Hn, Blds, m0, lane, acc);

#pragma unroll
    for (int j = 0; j < 4; ++j) {
        int n = 16 * j + (lane & 15);
#pragma unroll
        for (int r = 0; r < 8; ++r) {
            int m = m0 + r + 8 * hf;
            U[(size_t)m * 64 + n] = acc[j][r];
        }
    }
}

// ---------------------------------------------------------------------------
// Sequential scan: s_t = s_{t-1} @ A^T + u_t. One block, 4 waves; state
// (16 batches x 64) in LDS; A^T B-fragments live in registers for all T.
// ---------------------------------------------------------------------------
__global__ void __launch_bounds__(128) scan_kernel(const float* __restrict__ U,
                                                   const bf16_t* __restrict__ Abf,
                                                   bf16_t* __restrict__ Sout, int T) {
    __shared__ bf16_t s_lds[16 * 64];
    int lane = threadIdx.x & 31;
    int wave = threadIdx.x >> 5;
    int hf   = lane >> 4;
    int col  = wave * 16 + (lane & 15);

    for (int i = threadIdx.x; i < 16 * 64; i += 128) s_lds[i] = (bf16_t)0.0f;

    v16bf bA0 = load_b(Abf, 64, wave * 16, 0,  lane);
    v16bf bA1 = load_b(Abf, 64, wave * 16, 32, lane);
    __syncthreads();

    for (int t = 0; t < T; ++t) {
        v16bf a0 = load_a(s_lds, 64, 0, 0,  lane);
        v16bf a1 = load_a(s_lds, 64, 0, 32, lane);
        v8f acc = vzero8();
        acc = wmma_bf16(a0, bA0, acc);
        acc = wmma_bf16(a1, bA1, acc);
#pragma unroll
        for (int r = 0; r < 8; ++r) {
            int brow = r + 8 * hf;                       // batch index
            acc[r] += U[((size_t)brow * T + t) * 64 + col];
        }
        __syncthreads();
#pragma unroll
        for (int r = 0; r < 8; ++r) {
            int brow = r + 8 * hf;
            bf16_t v = (bf16_t)acc[r];
            s_lds[brow * 64 + col] = v;
            Sout[((size_t)brow * T + t) * 64 + col] = v;
        }
        __syncthreads();
    }
}

// ---------------------------------------------------------------------------
// Y[M x 512] = S[M x 64] @ Cm[512 x 64]^T + Hn[M x 512] @ Dm[512 x 512]^T,
// bf16 out. Dm slice staged in LDS; Cm (K=64) read direct from global.
// grid = (M/128, 8)
// ---------------------------------------------------------------------------
__global__ void __launch_bounds__(256) gemm_y(const bf16_t* __restrict__ S,
                                              const bf16_t* __restrict__ Cw,
                                              const bf16_t* __restrict__ Hn,
                                              const bf16_t* __restrict__ Dw,
                                              bf16_t* __restrict__ Y) {
    __shared__ bf16_t Blds[64 * 512];
    int lane  = threadIdx.x & 31;
    int wave  = threadIdx.x >> 5;
    int hf    = lane >> 4;
    int m0    = blockIdx.x * 128 + wave * 16;
    int n0blk = blockIdx.y * 64;

    stage_b512(Dw + (size_t)n0blk * 512, Blds);

    v8f acc[4];
#pragma unroll
    for (int j = 0; j < 4; ++j) acc[j] = vzero8();

    // s @ Cm^T (K = 64) from global while the Dm slice streams into LDS
#pragma unroll
    for (int k0 = 0; k0 < 64; k0 += 32) {
        v16bf a = load_a(S, 64, m0, k0, lane);
#pragma unroll
        for (int j = 0; j < 4; ++j) {
            v16bf b = load_b(Cw, 64, n0blk + 16 * j, k0, lane);
            acc[j] = wmma_bf16(a, b, acc[j]);
        }
    }

    stage_wait();
    gemm_core512(Hn, Blds, m0, lane, acc);

#pragma unroll
    for (int j = 0; j < 4; ++j) {
        int n = n0blk + 16 * j + (lane & 15);
#pragma unroll
        for (int r = 0; r < 8; ++r) {
            int m = m0 + r + 8 * hf;
            Y[(size_t)m * 512 + n] = (bf16_t)acc[j][r];
        }
    }
}

// ---------------------------------------------------------------------------
// Host side
// ---------------------------------------------------------------------------
extern "C" void kernel_launch(void* const* d_in, const int* in_sizes, int n_in,
                              void* d_out, int out_size, void* d_ws, size_t ws_size,
                              hipStream_t stream) {
    const int Bsz = 16, T = 4096, D = 512, DS = 64;
    const int M = Bsz * T;  // 65536

    const float* x     = (const float*)d_in[0];
    const float* A     = (const float*)d_in[1];
    const float* Bm    = (const float*)d_in[2];
    const float* Cm    = (const float*)d_in[3];
    const float* Dm    = (const float*)d_in[4];
    const float* Wi    = (const float*)d_in[5];
    const float* bi    = (const float*)d_in[6];
    const float* Wo    = (const float*)d_in[7];
    const float* bo    = (const float*)d_in[8];
    const float* gamma = (const float*)d_in[9];
    const float* beta  = (const float*)d_in[10];

    char* ws = (char*)d_ws;
    size_t off = 0;
    bf16_t* Xbf   = (bf16_t*)(ws + off); off += (size_t)M * D * 2;   // reused as Ybf
    bf16_t* Hnbf  = (bf16_t*)(ws + off); off += (size_t)M * D * 2;
    float*  U     = (float*)(ws + off);  off += (size_t)M * DS * 4;
    bf16_t* Sbf   = (bf16_t*)(ws + off); off += (size_t)M * DS * 2;
    bf16_t* Wi_bf = (bf16_t*)(ws + off); off += (size_t)D * D * 2;
    bf16_t* Wo_bf = (bf16_t*)(ws + off); off += (size_t)D * D * 2;
    bf16_t* Dm_bf = (bf16_t*)(ws + off); off += (size_t)D * D * 2;
    bf16_t* Bm_bf = (bf16_t*)(ws + off); off += (size_t)DS * D * 2;
    bf16_t* Cm_bf = (bf16_t*)(ws + off); off += (size_t)D * DS * 2;
    bf16_t* A_bf  = (bf16_t*)(ws + off); off += (size_t)DS * DS * 2;

    float* H   = (float*)d_out;   // scratch; fully overwritten by final GEMM
    float* Out = (float*)d_out;
    bf16_t* Ybf = Xbf;            // Xbf dead after first GEMM

    auto cvt = [&](const float* src, bf16_t* dst, int n) {
        cvt_bf16<<<(n + 255) / 256, 256, 0, stream>>>(src, dst, n);
    };

    cvt(x,  Xbf,   M * D);
    cvt(Wi, Wi_bf, D * D);
    cvt(Wo, Wo_bf, D * D);
    cvt(Dm, Dm_bf, D * D);
    cvt(Bm, Bm_bf, DS * D);
    cvt(Cm, Cm_bf, D * DS);
    cvt(A,  A_bf,  DS * DS);

    dim3 g512(M / 128, D / 64);
    // H = x @ Wi^T + bi
    gemm_n512<<<g512, 256, 0, stream>>>(Xbf, Wi_bf, bi, H);
    // Hn = LayerNorm(H)
    ln_rows<<<M, 256, 0, stream>>>(H, gamma, beta, Hnbf);
    // U = Hn @ Bm^T
    gemm_u<<<M / 128, 256, 0, stream>>>(Hnbf, Bm_bf, U);
    // sequential recurrence -> Sbf
    scan_kernel<<<1, 128, 0, stream>>>(U, A_bf, Sbf, T);
    // Y = S @ Cm^T + Hn @ Dm^T
    gemm_y<<<g512, 256, 0, stream>>>(Sbf, Cm_bf, Hnbf, Dm_bf, Ybf);
    // out = Y @ Wo^T + bo
    gemm_n512<<<g512, 256, 0, stream>>>(Ybf, Wo_bf, bo, Out);
}
